// RoPEMaskedAttentionHead_35046933135573
// MI455X (gfx1250) — matrix operand order repaired and checked by reading, not compiled, not run on hardware
//
#include <hip/hip_runtime.h>

// Problem constants (match reference)
#define BB 4
#define MM 4096
#define DD 512

typedef float  v8f    __attribute__((ext_vector_type(8)));
typedef __bf16 v8bf   __attribute__((ext_vector_type(8)));
typedef __bf16 v16bf  __attribute__((ext_vector_type(16)));
typedef int    i32x4  __attribute__((ext_vector_type(4)));
typedef int    i32x8  __attribute__((ext_vector_type(8)));

// ---------------------------------------------------------------------------
// helpers
// ---------------------------------------------------------------------------
static __device__ __forceinline__ void sched_fence() {
#if __has_builtin(__builtin_amdgcn_sched_barrier)
  __builtin_amdgcn_sched_barrier(0);   // keep preloaded tiles above the WMMAs
#endif
}

static __device__ __forceinline__ v16bf ld2x16B(const __bf16* p0, const __bf16* p1) {
  uint4 a = *(const uint4*)p0;             // 8 bf16
  uint4 b = *(const uint4*)p1;             // 8 bf16
  v8bf lo = __builtin_bit_cast(v8bf, a);
  v8bf hi = __builtin_bit_cast(v8bf, b);
  return __builtin_shufflevector(lo, hi, 0,1,2,3,4,5,6,7,8,9,10,11,12,13,14,15);
}

static __device__ __forceinline__ v8bf cvt8(const float* p) {
  float4 a = ((const float4*)p)[0];
  float4 b = ((const float4*)p)[1];
  v8bf r;
  r[0]=(__bf16)a.x; r[1]=(__bf16)a.y; r[2]=(__bf16)a.z; r[3]=(__bf16)a.w;
  r[4]=(__bf16)b.x; r[5]=(__bf16)b.y; r[6]=(__bf16)b.z; r[7]=(__bf16)b.w;
  return r;
}

static __device__ __forceinline__ v8f wmma_bf16(v16bf A, v16bf B, v8f C) {
  // D = A(16x32 bf16) x B(32x16 bf16) + C(16x16 f32)
  return __builtin_amdgcn_wmma_f32_16x16x32_bf16(false, A, false, B, (short)0, C,
                                                 false, false);
}

// TDM: DMA one 32x512 bf16 tile (32 KB) from global (row-major, row stride 512)
// into LDS.  D# group0/group1 per CDNA5 ISA §8; 2-D tensor -> groups 2/3 NULL.
// OOB rows (beyond tensor_dim1) read as zero per spec.
static __device__ __forceinline__ void tdm_load_ktile(const __bf16* src,
                                                      __bf16* lds_dst) {
  unsigned long long ga = (unsigned long long)(const void*)src;
  i32x4 g0;
  g0.x = 1;                                           // count=1 (valid user D#)
  g0.y = (int)(unsigned)(unsigned long long)(void*)lds_dst;  // lds_addr
  g0.z = (int)(unsigned)ga;                           // global_addr[31:0]
  g0.w = (int)((unsigned)(ga >> 32) | 0x80000000u);   // addr[56:32] | type=2
  i32x8 g1;
  g1.s0 = 0x00010000;          // workgroup_mask=0, data_size=1 (2 bytes)
  g1.s1 = (int)(512u << 16);   // tensor_dim0 = 512   (bits [79:48])
  g1.s2 = (int)(4096u << 16);  // tensor_dim1 = 4096  (bits [111:80])
  g1.s3 = (int)(512u << 16);   // tile_dim0 = 512     (bits [127:112])
  g1.s4 = 32;                  // tile_dim1 = 32      (bits [143:128])
  g1.s5 = 512;                 // tensor_dim0_stride = 512 (bits [207:160])
  g1.s6 = 0;
  g1.s7 = 0;
  asm volatile("tensor_load_to_lds %0, %1" :: "s"(g0), "s"(g1) : "memory");
}

// ---------------------------------------------------------------------------
// K1: convert W_q / W_k (f32 row-major [k][n]) -> bf16 transposed [n][k]
// ---------------------------------------------------------------------------
__global__ void wt_convert_kernel(const float* __restrict__ Wq,
                                  const float* __restrict__ Wk,
                                  __bf16* __restrict__ Wtq,
                                  __bf16* __restrict__ Wtk) {
  const int n = blockIdx.x;
  const float* W  = blockIdx.y ? Wk  : Wq;
  __bf16*      Wt = blockIdx.y ? Wtk : Wtq;
  for (int k = threadIdx.x; k < DD; k += blockDim.x)
    Wt[(size_t)n * DD + k] = (__bf16)W[(size_t)k * DD + n];
}

// ---------------------------------------------------------------------------
// K2: s[b*M+m] = x[b,m,:] . u_v   (one wave per row)
// ---------------------------------------------------------------------------
__global__ void rowdot_kernel(const float* __restrict__ x,
                              const float* __restrict__ u,
                              float* __restrict__ s) {
  const int w = threadIdx.x >> 5, l = threadIdx.x & 31;
  const size_t r = (size_t)blockIdx.x * 8 + w;
  const float* xr = x + r * DD;
  float sum = 0.f;
  #pragma unroll
  for (int i = 0; i < DD / 32; ++i) sum += xr[l + i * 32] * u[l + i * 32];
  #pragma unroll
  for (int d = 16; d >= 1; d >>= 1) sum += __shfl_xor(sum, d, 32);
  if (l == 0) s[r] = sum;
}

// ---------------------------------------------------------------------------
// K3: vT[b][d][m] = s[b,m] * v_v[d]   (bf16, transposed for PV B-tiles)
// ---------------------------------------------------------------------------
__global__ void vtfill_kernel(const float* __restrict__ s,
                              const float* __restrict__ vv,
                              __bf16* __restrict__ vT) {
  const int row = blockIdx.x;                 // b*512 + d
  const int b = row >> 9, d = row & (DD - 1);
  const float vd = vv[d];
  const float* sb = s + (size_t)b * MM;
  __bf16* o = vT + (size_t)row * MM;
  for (int m = threadIdx.x; m < MM; m += blockDim.x)
    o[m] = (__bf16)(sb[m] * vd);
}

// ---------------------------------------------------------------------------
// K4: q/k = x@W (WMMA bf16), RoPE, store bf16.  blockIdx.y: 0->q, 1->k
//     All 8 B-tiles of a K-step preloaded; sched_barrier keeps them above the
//     WMMA chain so VMEM latency overlaps the matrix pipe.
// ---------------------------------------------------------------------------
__global__ __launch_bounds__(128) void proj_rope_kernel(
    const float* __restrict__ x,
    const __bf16* __restrict__ Wtq, const __bf16* __restrict__ Wtk,
    __bf16* __restrict__ qr, __bf16* __restrict__ kr) {
  const __bf16* Wt = blockIdx.y ? Wtk : Wtq;
  __bf16*      outp = blockIdx.y ? kr  : qr;

  const int tid = threadIdx.x;
  const int w = tid >> 5, l = tid & 31, lh = l >> 4, ln = l & 15;
  const int rt = blockIdx.x;
  const int m0 = (rt & 255) << 4;
  const size_t r0 = (size_t)rt * 16;

  v8f acc[8] = {};

  for (int ks = 0; ks < DD / 32; ++ks) {
    const int kb = ks * 32;
    // A-tile from x (f32 -> bf16)
    const float* xr = x + (r0 + ln) * DD + kb;
    v8bf lo = cvt8(xr + lh * 8);
    v8bf hi = cvt8(xr + 16 + lh * 8);
    v16bf A = __builtin_shufflevector(lo, hi, 0,1,2,3,4,5,6,7,8,9,10,11,12,13,14,15);
    // Preload all 8 B-tiles (one clause); fence; then the WMMA chain waits
    // only progressively (loadcnt <= 14,12,...) instead of 0 per tile.
    v16bf Bt[8];
    #pragma unroll
    for (int t = 0; t < 8; ++t) {
      const __bf16* wr = Wt + (size_t)(w * 128 + t * 16 + ln) * DD + kb;
      Bt[t] = ld2x16B(wr + lh * 8, wr + 16 + lh * 8);
    }
    sched_fence();
    #pragma unroll
    for (int t = 0; t < 8; ++t) acc[t] = wmma_bf16(A, Bt[t], acc[t]);
  }

  // RoPE: theta_h = 10000^(-2(h-1)/512) = exp(-0.03597789*(h-1))
  #pragma unroll
  for (int t = 0; t < 8; ++t) {
    const int c = w * 128 + t * 16 + ln;
    const int h = c >> 1;
    const float theta = __expf(-0.035977891f * ((float)h - 1.0f));
    #pragma unroll
    for (int j = 0; j < 8; ++j) {
      const int m = m0 + j + 8 * lh;
      float val = acc[t][j];
      float other = __shfl_xor(val, 1, 32);
      float sn, cs;
      __sincosf((float)m * theta, &sn, &cs);
      float o = (c & 1) ? (-other * sn + val * cs) : (val * cs + other * sn);
      outp[(r0 + j + 8 * lh) * DD + c] = (__bf16)o;
    }
  }
}

// ---------------------------------------------------------------------------
// K5: causal flash attention.  1 block = 16 queries, 4 waves; wave w owns
//     output dims [w*128,+128).  Keys in tiles of 32, DMA'd into LDS by the
//     Tensor Data Mover.  The TDM for tile kt+1 is issued right after the
//     score-reduction barrier (which proves all waves finished reading the
//     K-tile), so the DMA overlaps softmax + PV; the wait sits at the top of
//     the next iteration.  2 barriers per key tile, 44 KB LDS.
// ---------------------------------------------------------------------------
__global__ __launch_bounds__(128) void attn_kernel(
    const __bf16* __restrict__ qr, const __bf16* __restrict__ kr,
    const __bf16* __restrict__ vT, float* __restrict__ out) {
  __shared__ __align__(16) __bf16 krt[32 * DD];          // 32 KB K-tile (TDM dest)
  __shared__ float red[4][2][256];                       // 8 KB score reduce
  __shared__ __align__(16) __bf16 pst[4][16 * 32];       // 4 KB P repack

  const int tid = threadIdx.x;
  const int w = tid >> 5, l = tid & 31, lh = l >> 4, ln = l & 15;

  const int qt = blockIdx.x;
  const int b  = qt >> 8;
  const int q0 = (qt & 255) << 4;
  const size_t rowbase = (size_t)b * MM + q0;

  const int   ktiles = (q0 + 15) / 32 + 1;
  const float scale  = 0.044194173824159216f;  // 1/sqrt(512)

  // Prologue: start DMA of K-tile 0 while we fetch the Q tiles.
  if (threadIdx.x < 32)
    tdm_load_ktile(kr + (size_t)b * MM * DD, krt);

  // Preload qr A-tiles for this wave's 128-dim chunk (4 K-steps of 32)
  v16bf qA[4];
  {
    const __bf16* qrow = qr + (rowbase + ln) * DD + w * 128;
    #pragma unroll
    for (int ks = 0; ks < 4; ++ks) {
      const __bf16* p = qrow + ks * 32;
      qA[ks] = ld2x16B(p + lh * 8, p + 16 + lh * 8);
    }
  }

  v8f acc[8] = {};
  float mrow[8], ssum[8];
  #pragma unroll
  for (int j = 0; j < 8; ++j) { mrow[j] = -__builtin_inff(); ssum[j] = 0.f; }

  for (int kt = 0; kt < ktiles; ++kt) {
    const int k0 = kt * 32;

    // ---- wait for the in-flight K-tile DMA, then make it visible
    if (threadIdx.x < 32) __builtin_amdgcn_s_wait_tensorcnt(0);
    __syncthreads();      // krt ready; also fences red reads (kt-1) vs writes

    if (kt + 1 < ktiles) {                        // warm next V tile toward L2
      __builtin_prefetch(vT + ((size_t)b * DD + w * 128 + (l & 7) * 16) * MM
                            + k0 + 32, 0, 1);
    }

    // ---- partial scores over dims [w*128,+128): preload 8 LDS B-tiles,
    //      fence, then run the WMMA chain (ds waits overlap matrix pipe)
    v16bf Bs[8];
    #pragma unroll
    for (int nh = 0; nh < 2; ++nh) {
      const __bf16* krow = &krt[(size_t)(nh * 16 + ln) * DD + w * 128];
      #pragma unroll
      for (int ks = 0; ks < 4; ++ks)
        Bs[nh * 4 + ks] = ld2x16B(krow + ks * 32 + lh * 8, krow + ks * 32 + 16 + lh * 8);
    }
    sched_fence();
    v8f c0 = {}, c1 = {};
    #pragma unroll
    for (int ks = 0; ks < 4; ++ks) c0 = wmma_bf16(qA[ks], Bs[ks], c0);
    #pragma unroll
    for (int ks = 0; ks < 4; ++ks) c1 = wmma_bf16(qA[ks], Bs[4 + ks], c1);

    // ---- cross-wave reduction
    #pragma unroll
    for (int j = 0; j < 8; ++j) {
      const int idx = (j + 8 * lh) * 16 + ln;
      red[w][0][idx] = c0[j];
      red[w][1][idx] = c1[j];
    }
    __syncthreads();      // all waves done reading krt AND writing red

    // ---- kick off the next K-tile DMA; it overlaps softmax + PV below
    if (kt + 1 < ktiles && threadIdx.x < 32)
      tdm_load_ktile(kr + ((size_t)b * MM + k0 + 32) * DD, krt);

    // ---- online softmax (redundant per wave; identical results)
    float p0[8], p1[8], alpha[8];
    #pragma unroll
    for (int j = 0; j < 8; ++j) {
      const int idx = (j + 8 * lh) * 16 + ln;
      float v0 = (red[0][0][idx] + red[1][0][idx]) + (red[2][0][idx] + red[3][0][idx]);
      float v1 = (red[0][1][idx] + red[1][1][idx]) + (red[2][1][idx] + red[3][1][idx]);
      v0 *= scale; v1 *= scale;
      const int row = q0 + j + 8 * lh;
      if (k0 + ln > row)      v0 = -__builtin_inff();
      if (k0 + 16 + ln > row) v1 = -__builtin_inff();
      float t = fmaxf(v0, v1);
      #pragma unroll
      for (int d = 8; d >= 1; d >>= 1) t = fmaxf(t, __shfl_xor(t, d, 32));
      const float nm = fmaxf(mrow[j], t);
      alpha[j] = __expf(mrow[j] - nm);
      p0[j] = __expf(v0 - nm);
      p1[j] = __expf(v1 - nm);
      float ts = p0[j] + p1[j];
      #pragma unroll
      for (int d = 8; d >= 1; d >>= 1) ts += __shfl_xor(ts, d, 32);
      ssum[j] = ssum[j] * alpha[j] + ts;
      mrow[j] = nm;
    }

    // ---- rescale accumulator
    #pragma unroll
    for (int t = 0; t < 8; ++t)
      #pragma unroll
      for (int j = 0; j < 8; ++j) acc[t][j] *= alpha[j];

    // ---- repack P (C layout -> A layout) through per-wave LDS staging
    #pragma unroll
    for (int j = 0; j < 8; ++j) {
      pst[w][(j + 8 * lh) * 32 + ln]      = (__bf16)p0[j];
      pst[w][(j + 8 * lh) * 32 + 16 + ln] = (__bf16)p1[j];
    }
    const __bf16* pp = &pst[w][ln * 32];           // same-wave DS is in-order
    v16bf Pa = ld2x16B(pp + lh * 8, pp + 16 + lh * 8);

    // ---- PV: preload 8 V B-tiles (global, L2-resident), fence, WMMA chain
    v16bf Bv[8];
    #pragma unroll
    for (int t = 0; t < 8; ++t) {
      const __bf16* vrow =
          vT + ((size_t)b * DD + w * 128 + t * 16 + ln) * MM + k0;
      Bv[t] = ld2x16B(vrow + lh * 8, vrow + 16 + lh * 8);
    }
    sched_fence();
    #pragma unroll
    for (int t = 0; t < 8; ++t) acc[t] = wmma_bf16(Pa, Bv[t], acc[t]);
  }

  // ---- epilogue: divide by row sums, store f32
  #pragma unroll
  for (int j = 0; j < 8; ++j) {
    const float inv = 1.0f / ssum[j];
    const int row = q0 + j + 8 * lh;
    float* orow = out + ((size_t)b * MM + row) * DD + w * 128;
    #pragma unroll
    for (int t = 0; t < 8; ++t) orow[t * 16 + ln] = acc[t][j] * inv;
  }
}

// ---------------------------------------------------------------------------
// launch
// ---------------------------------------------------------------------------
extern "C" void kernel_launch(void* const* d_in, const int* in_sizes, int n_in,
                              void* d_out, int out_size, void* d_ws, size_t ws_size,
                              hipStream_t stream) {
  const float* x  = (const float*)d_in[0];
  const float* Wq = (const float*)d_in[1];
  const float* Wk = (const float*)d_in[2];
  const float* uv = (const float*)d_in[3];
  const float* vv = (const float*)d_in[4];
  float* out = (float*)d_out;

  char* ws = (char*)d_ws;
  const size_t QK_BYTES = (size_t)BB * MM * DD * 2;        // 16 MB each
  __bf16* qr  = (__bf16*)(ws);
  __bf16* kr  = (__bf16*)(ws + QK_BYTES);
  __bf16* vT  = (__bf16*)(ws + 2 * QK_BYTES);
  __bf16* Wtq = (__bf16*)(ws + 3 * QK_BYTES);
  __bf16* Wtk = (__bf16*)(ws + 3 * QK_BYTES + (size_t)DD * DD * 2);
  float*  s   = (float*) (ws + 3 * QK_BYTES + 2 * (size_t)DD * DD * 2);

  wt_convert_kernel<<<dim3(DD, 2), 256, 0, stream>>>(Wq, Wk, Wtq, Wtk);
  rowdot_kernel<<<(BB * MM) / 8, 256, 0, stream>>>(x, uv, s);
  vtfill_kernel<<<BB * DD, 256, 0, stream>>>(s, vv, vT);
  proj_rope_kernel<<<dim3((BB * MM) / 16, 2), 128, 0, stream>>>(x, Wtq, Wtk, qr, kr);
  attn_kernel<<<(BB * MM) / 16, 128, 0, stream>>>(qr, kr, vT, out);
}
